// FastSelfAttention_51917564674491
// MI455X (gfx1250) — matrix operand order repaired and checked by reading, not compiled, not run on hardware
//
#include <hip/hip_runtime.h>
#include <hip/hip_bf16.h>

// FastSelfAttention for MI455X (gfx1250, wave32, WMMA + async LDS staging).
// B=4, S=4096, D=1024, H=16, DH=64. M = B*S = 16384.

typedef __attribute__((ext_vector_type(16))) __bf16 v16bf;
typedef __attribute__((ext_vector_type(8)))  float  v8f;
typedef __attribute__((ext_vector_type(4)))  int    v4i;

#define TILE_M 128
#define TILE_N 256
#define TILE_K 32

// ---------------------------------------------------------------------------
// async memory->LDS helpers (CDNA5 GLOBAL_LOAD_ASYNC_TO_LDS_B128, ASYNCcnt)
// ---------------------------------------------------------------------------
#if __has_builtin(__builtin_amdgcn_global_load_async_to_lds_b128) && \
    __has_builtin(__builtin_amdgcn_s_wait_asynccnt)
#define ATH_ASYNC_LDS 1
#else
#define ATH_ASYNC_LDS 0
#endif

__device__ __forceinline__ void async_copy16(const __bf16* gsrc, __bf16* ldst) {
#if ATH_ASYNC_LDS
  __builtin_amdgcn_global_load_async_to_lds_b128(
      (__attribute__((address_space(1))) v4i*)gsrc,
      (__attribute__((address_space(3))) v4i*)ldst, 0, 0);
#else
  *(uint4*)ldst = *(const uint4*)gsrc;
#endif
}

__device__ __forceinline__ void wait_async_lds() {
#if ATH_ASYNC_LDS
  __builtin_amdgcn_s_wait_asynccnt(0);
#else
  asm volatile("" ::: "memory");
#endif
}

// ---------------------------------------------------------------------------
// fp32 -> bf16 convert (plain layout)
// ---------------------------------------------------------------------------
__global__ __launch_bounds__(256) void f32_to_bf16_kernel(
    const float* __restrict__ in, unsigned short* __restrict__ out_, size_t n) {
  __bf16* out = (__bf16*)out_;
  size_t i = (size_t)blockIdx.x * blockDim.x + threadIdx.x;
  if (i < n) out[i] = (__bf16)in[i];
}

// ---------------------------------------------------------------------------
// fp32 [K,N] -> bf16 [N,K] transposed convert (weights, one-time)
// ---------------------------------------------------------------------------
__global__ __launch_bounds__(256) void transpose_f32_to_bf16_kernel(
    const float* __restrict__ in, unsigned short* __restrict__ out_, int K, int N) {
  __bf16* out = (__bf16*)out_;
  size_t i = (size_t)blockIdx.x * blockDim.x + threadIdx.x;
  if (i < (size_t)K * N) {
    int k = (int)(i / N);
    int n = (int)(i % N);
    out[(size_t)n * K + k] = (__bf16)in[i];
  }
}

// ---------------------------------------------------------------------------
// bf16 WMMA GEMM: C[M,N] = A[M,K] * Bt[N,K]^T + bias[N] (+ residual[M,N])
// Block tile 128x256, 8 waves (2x4), each wave 64x64 via 16x 16x16x32 WMMA.
// A and Bt are staged to LDS with async b128 copies, double-buffered.
// ---------------------------------------------------------------------------
__global__ __launch_bounds__(256) void gemm_bf16_wmma_kernel(
    const unsigned short* __restrict__ A_,   // [M,K] bf16 row-major
    const unsigned short* __restrict__ Bt_,  // [N,K] bf16 row-major (= B^T)
    const float* __restrict__ bias,          // [N]
    const float* __restrict__ residual,      // [M,N] or nullptr
    float* __restrict__ C,                   // [M,N]
    int M, int N, int K) {
  const __bf16* A  = (const __bf16*)A_;
  const __bf16* Bt = (const __bf16*)Bt_;

  __shared__ __bf16 As[2][TILE_M * TILE_K];  // [m][k], k contiguous
  __shared__ __bf16 Bs[2][TILE_N * TILE_K];  // [n][k], k contiguous

  const int tid  = threadIdx.x;
  const int lane = tid & 31;
  const int wid  = tid >> 5;   // 0..7
  const int wm   = wid >> 2;   // 0..1  (M direction)
  const int wn   = wid & 3;    // 0..3  (N direction)
  const int bm   = blockIdx.y * TILE_M;
  const int bn   = blockIdx.x * TILE_N;

  const int lrow = lane & 15;
  const int lhi  = lane >> 4;  // 0 or 1

  v8f acc[4][4] = {};

  // stage one K-tile (k0) into buffer `buf`: branchless, 2+4 chunks/thread
  auto stage = [&](int buf, int k0) {
    #pragma unroll
    for (int it = 0; it < (TILE_M * TILE_K / 8) / 256; ++it) {   // 2 iters
      int idx = tid + it * 256;
      int r   = idx >> 2;          // row 0..127
      int c8  = idx & 3;           // 8-elem chunk in k
      async_copy16(&A[(size_t)(bm + r) * K + k0 + c8 * 8],
                   &As[buf][r * TILE_K + c8 * 8]);
    }
    #pragma unroll
    for (int it = 0; it < (TILE_N * TILE_K / 8) / 256; ++it) {   // 4 iters
      int idx = tid + it * 256;
      int n   = idx >> 2;          // row 0..255
      int c8  = idx & 3;
      async_copy16(&Bt[(size_t)(bn + n) * K + k0 + c8 * 8],
                   &Bs[buf][n * TILE_K + c8 * 8]);
    }
  };

  const int ntiles = K / TILE_K;
  stage(0, 0);
  wait_async_lds();
  __syncthreads();

  for (int t = 0; t < ntiles; ++t) {
    const int cur = t & 1;
    if (t + 1 < ntiles) stage(cur ^ 1, (t + 1) * TILE_K);

    const __bf16* as = As[cur];
    const __bf16* bs = Bs[cur];

    // fragments (layouts per CDNA5 ISA 7.12.2):
    // A 16x32: lane holds K = 8*lhi..+7 and 16+8*lhi..+7 of row (lane&15)
    // B 32x16: lane holds K = 8*lhi..+7 and 16+8*lhi..+7 of column (lane&15)
    union FragU { v16bf v; uint4 u[2]; };
    FragU afr[4], bfr[4];
    #pragma unroll
    for (int i = 0; i < 4; ++i) {
      int r = wm * 64 + i * 16 + lrow;
      afr[i].u[0] = *(const uint4*)&as[r * TILE_K + lhi * 8];
      afr[i].u[1] = *(const uint4*)&as[r * TILE_K + 16 + lhi * 8];
    }
    #pragma unroll
    for (int j = 0; j < 4; ++j) {
      int c = wn * 64 + j * 16 + lrow;
      bfr[j].u[0] = *(const uint4*)&bs[c * TILE_K + lhi * 8];
      bfr[j].u[1] = *(const uint4*)&bs[c * TILE_K + 16 + lhi * 8];
    }

    #pragma unroll
    for (int i = 0; i < 4; ++i)
      #pragma unroll
      for (int j = 0; j < 4; ++j)
        acc[i][j] = __builtin_amdgcn_wmma_f32_16x16x32_bf16(
            false, afr[i].v, false, bfr[j].v, (short)0, acc[i][j], false, false);

    wait_async_lds();     // next tile fully landed in LDS
    __syncthreads();      // all waves done reading `cur` / writing `cur^1`
  }

  // epilogue: C/D layout: M = vgpr + 8*lhi, N = lane&15
  #pragma unroll
  for (int i = 0; i < 4; ++i) {
    #pragma unroll
    for (int j = 0; j < 4; ++j) {
      int col = bn + wn * 64 + j * 16 + lrow;
      float bv = bias ? bias[col] : 0.0f;
      #pragma unroll
      for (int r = 0; r < 8; ++r) {
        int row = bm + wm * 64 + i * 16 + lhi * 8 + r;
        size_t o = (size_t)row * N + col;
        float v = acc[i][j][r] + bv;
        if (residual) v += residual[o];
        C[o] = v;
      }
    }
  }
}

// ---------------------------------------------------------------------------
// score = (X[M,1024] @ W[1024,16] + bias[16]) * scale   -> S_out[M,16]
// ---------------------------------------------------------------------------
__global__ __launch_bounds__(256) void score_kernel(
    const float* __restrict__ X, const float* __restrict__ W,
    const float* __restrict__ bias, float scale,
    float* __restrict__ S_out) {
  __shared__ float row[1024];
  __shared__ float part[16][17];
  const int m   = blockIdx.x;
  const int tid = threadIdx.x;
  #pragma unroll
  for (int j = tid; j < 1024; j += 256) row[j] = X[(size_t)m * 1024 + j];
  __syncthreads();
  const int h = tid & 15;
  const int p = tid >> 4;
  float s = 0.f;
  #pragma unroll 8
  for (int j = p * 64; j < p * 64 + 64; ++j) s += row[j] * W[j * 16 + h];
  part[p][h] = s;
  __syncthreads();
  if (tid < 16) {
    float t = 0.f;
    #pragma unroll
    for (int q = 0; q < 16; ++q) t += part[q][tid];
    S_out[(size_t)m * 16 + tid] = (t + bias[tid]) * scale;
  }
}

// ---------------------------------------------------------------------------
// softmax over S per (b,h): X,Y are [B,S,H]; one workgroup per (b,h)
// ---------------------------------------------------------------------------
__global__ __launch_bounds__(256) void softmax_seq_kernel(
    const float* __restrict__ X, float* __restrict__ Y, int S, int H) {
  const int b = blockIdx.x / H;
  const int h = blockIdx.x % H;
  const size_t base = (size_t)b * S * H + h;
  __shared__ float red[256];
  const int tid = threadIdx.x;

  float mx = -3.4e38f;
  for (int s = tid; s < S; s += 256) mx = fmaxf(mx, X[base + (size_t)s * H]);
  red[tid] = mx;
  __syncthreads();
  for (int off = 128; off > 0; off >>= 1) {
    if (tid < off) red[tid] = fmaxf(red[tid], red[tid + off]);
    __syncthreads();
  }
  const float m = red[0];
  __syncthreads();

  float sum = 0.f;
  for (int s = tid; s < S; s += 256) sum += __expf(X[base + (size_t)s * H] - m);
  red[tid] = sum;
  __syncthreads();
  for (int off = 128; off > 0; off >>= 1) {
    if (tid < off) red[tid] += red[tid + off];
    __syncthreads();
  }
  const float inv = 1.0f / red[0];
  for (int s = tid; s < S; s += 256)
    Y[base + (size_t)s * H] = __expf(X[base + (size_t)s * H] - m) * inv;
}

// ---------------------------------------------------------------------------
// pooled[b, h*64+d] = sum_s Wgt[b,s,h] * X[b,s,h*64+d]; one WG per (b,h)
// ---------------------------------------------------------------------------
__global__ __launch_bounds__(256) void pool_kernel(
    const float* __restrict__ Wgt,  // [B,S,H]
    const float* __restrict__ X,    // [B,S,D]
    float* __restrict__ P,          // [B,D]
    int S, int H, int DH, int D) {
  const int b = blockIdx.x / H;
  const int h = blockIdx.x % H;
  const int d  = threadIdx.x & 63;
  const int sp = threadIdx.x >> 6;   // 0..3
  __shared__ float red[4][64];
  float acc = 0.f;
  for (int s = sp; s < S; s += 4) {
    float wv = Wgt[((size_t)b * S + s) * H + h];
    acc += wv * X[((size_t)b * S + s) * D + h * DH + d];
  }
  red[sp][d] = acc;
  __syncthreads();
  if (sp == 0)
    P[(size_t)b * D + h * DH + d] = red[0][d] + red[1][d] + red[2][d] + red[3][d];
}

// ---------------------------------------------------------------------------
// elementwise gates
// ---------------------------------------------------------------------------
__global__ __launch_bounds__(256) void gate_mul_f32_kernel(
    const float* __restrict__ X, const float* __restrict__ P,
    float* __restrict__ Y, int D, size_t SD, size_t n) {
  size_t i = (size_t)blockIdx.x * blockDim.x + threadIdx.x;
  if (i < n) {
    size_t b = i / SD;
    int d = (int)(i % (size_t)D);
    Y[i] = X[i] * P[b * (size_t)D + d];
  }
}

__global__ __launch_bounds__(256) void gate_mul_bf16_kernel(
    const float* __restrict__ X, const float* __restrict__ P,
    unsigned short* __restrict__ Y_, int D, size_t SD, size_t n) {
  __bf16* Y = (__bf16*)Y_;
  size_t i = (size_t)blockIdx.x * blockDim.x + threadIdx.x;
  if (i < n) {
    size_t b = i / SD;
    int d = (int)(i % (size_t)D);
    Y[i] = (__bf16)(X[i] * P[b * (size_t)D + d]);
  }
}

// ---------------------------------------------------------------------------
// launch
// ---------------------------------------------------------------------------
extern "C" void kernel_launch(void* const* d_in, const int* in_sizes, int n_in,
                              void* d_out, int out_size, void* d_ws, size_t ws_size,
                              hipStream_t stream) {
  (void)in_sizes; (void)n_in; (void)out_size; (void)ws_size;
  constexpr int Bb = 4, S = 4096, D = 1024, H = 16, DH = 64;
  constexpr int M = Bb * S;                   // 16384
  constexpr size_t MD = (size_t)M * D;        // 16.7M
  const float scale = 0.125f;                 // 1/sqrt(64)

  const float* hidden = (const float*)d_in[0];
  const float* Wq  = (const float*)d_in[1];
  const float* bq  = (const float*)d_in[2];
  const float* Wqa = (const float*)d_in[3];
  const float* bqa = (const float*)d_in[4];
  const float* Wk  = (const float*)d_in[5];
  const float* bk  = (const float*)d_in[6];
  const float* Wka = (const float*)d_in[7];
  const float* bka = (const float*)d_in[8];
  const float* Wt  = (const float*)d_in[9];
  const float* bt  = (const float*)d_in[10];
  float* out = (float*)d_out;

  // workspace layout
  char* w = (char*)d_ws;
  unsigned short* hbf  = (unsigned short*)w; w += MD * 2;            // hidden bf16 (reused for wv bf16)
  unsigned short* wqbT = (unsigned short*)w; w += (size_t)D * D * 2; // Wq^T bf16 [N,K]
  unsigned short* wkbT = (unsigned short*)w; w += (size_t)D * D * 2;
  unsigned short* wtbT = (unsigned short*)w; w += (size_t)D * D * 2;
  float* mq     = (float*)w; w += MD * 4;                            // mixed_q
  float* mk     = (float*)w; w += MD * 4;                            // mixed_k -> mixed_qk (in place)
  float* score  = (float*)w; w += (size_t)M * H * 4;
  float* smw    = (float*)w; w += (size_t)M * H * 4;
  float* pooled = (float*)w; w += (size_t)Bb * D * 4;
  unsigned short* wvb = hbf;                                         // reuse

  const int cvt_blocks_big = (int)((MD + 255) / 256);
  const int cvt_blocks_w   = (int)(((size_t)D * D + 255) / 256);

  // 1. bf16 converts (weights transposed to [N,K] for contiguous async B tiles)
  f32_to_bf16_kernel<<<cvt_blocks_big, 256, 0, stream>>>(hidden, hbf, MD);
  transpose_f32_to_bf16_kernel<<<cvt_blocks_w, 256, 0, stream>>>(Wq, wqbT, D, D);
  transpose_f32_to_bf16_kernel<<<cvt_blocks_w, 256, 0, stream>>>(Wk, wkbT, D, D);
  transpose_f32_to_bf16_kernel<<<cvt_blocks_w, 256, 0, stream>>>(Wt, wtbT, D, D);

  dim3 ggrid(D / TILE_N, M / TILE_M);   // (4, 128)

  // 2. mixed_q = hidden @ Wq + bq ; mixed_k = hidden @ Wk + bk
  gemm_bf16_wmma_kernel<<<ggrid, 256, 0, stream>>>(hbf, wqbT, bq, nullptr, mq, M, D, D);
  gemm_bf16_wmma_kernel<<<ggrid, 256, 0, stream>>>(hbf, wkbT, bk, nullptr, mk, M, D, D);

  // 3. q_score -> softmax -> pooled_q
  score_kernel<<<M, 256, 0, stream>>>(mq, Wqa, bqa, scale, score);
  softmax_seq_kernel<<<Bb * H, 256, 0, stream>>>(score, smw, S, H);
  pool_kernel<<<Bb * H, 256, 0, stream>>>(smw, mq, pooled, S, H, DH, D);

  // 4. mixed_qk = mixed_k * pooled_q (in place on mk)
  gate_mul_f32_kernel<<<cvt_blocks_big, 256, 0, stream>>>(mk, pooled, mk, D, (size_t)S * D, MD);

  // 5. k_score -> softmax -> pooled_k
  score_kernel<<<M, 256, 0, stream>>>(mk, Wka, bka, scale, score);
  softmax_seq_kernel<<<Bb * H, 256, 0, stream>>>(score, smw, S, H);
  pool_kernel<<<Bb * H, 256, 0, stream>>>(smw, mk, pooled, S, H, DH, D);

  // 6. wv = pooled_k * mixed_q  (bf16, reusing hidden-bf16 buffer)
  gate_mul_bf16_kernel<<<cvt_blocks_big, 256, 0, stream>>>(mq, pooled, wvb, D, (size_t)S * D, MD);

  // 7. out = wv @ Wt + bt + mixed_q
  gemm_bf16_wmma_kernel<<<ggrid, 256, 0, stream>>>(wvb, wtbT, bt, mq, out, M, D, D);
}
